// Model_39676907886085
// MI455X (gfx1250) — matrix operand order repaired and checked by reading, not compiled, not run on hardware
//
#include <hip/hip_runtime.h>

typedef __bf16 bf16x16 __attribute__((ext_vector_type(16)));
typedef __bf16 bf16x4  __attribute__((ext_vector_type(4)));
typedef float  f32x8   __attribute__((ext_vector_type(8)));
typedef float  f32x4   __attribute__((ext_vector_type(4)));

static constexpr int kLQ  = 2048;
static constexpr int kLK  = 2048;
static constexpr int kDH  = 64;
static constexpr int kNH  = 16;
static constexpr int kKT  = 32;   // k-tile depth per iteration
static constexpr int kKSb = 72;   // LDS row stride, K tile (bf16; 144B, 16B-aligned, bank-clean)
static constexpr int kVSb = 40;   // LDS row stride, V^T tile (bf16; 80B, 16B-aligned, bank-clean)
static constexpr int kPS  = 40;   // LDS row stride, P stage / mask tile (bf16)

#define WMMA_BF16(A, B, C) \
  __builtin_amdgcn_wmma_f32_16x16x32_bf16(false, (A), false, (B), (short)0, (C), false, false)

// raw VALU max (avoids clang's NaN-canonicalize companion ops)
#define VMAXF(a, b)                                                        \
  ({ float _d;                                                             \
     asm("v_max_num_f32 %0, %1, %2" : "=v"(_d) : "v"(a), "v"(b));          \
     _d; })

// fused single-instruction DPP ring-reduction steps (VALU pipe, no LDS)
#define ROR_MAX(v, N)                                                      \
  ({ float _d;                                                             \
     asm("v_max_num_f32_dpp %0, %1, %1 row_ror:" #N                        \
         " row_mask:0xf bank_mask:0xf bound_ctrl:1"                        \
         : "=v"(_d) : "v"(v));                                             \
     _d; })
#define ROR_ADD(v, N)                                                      \
  ({ float _d;                                                             \
     asm("v_add_f32_dpp %0, %1, %1 row_ror:" #N                            \
         " row_mask:0xf bank_mask:0xf bound_ctrl:1"                        \
         : "=v"(_d) : "v"(v));                                             \
     _d; })

__device__ __forceinline__ float red_max16(float v) {
  v = ROR_MAX(v, 1); v = ROR_MAX(v, 2); v = ROR_MAX(v, 4); v = ROR_MAX(v, 8);
  return v;
}
__device__ __forceinline__ float red_sum16(float v) {
  v = ROR_ADD(v, 1); v = ROR_ADD(v, 2); v = ROR_ADD(v, 4); v = ROR_ADD(v, 8);
  return v;
}

// WMMA B-fragment (32x16 bf16): lane = col, elem e -> k = 16*lg + e
union BFrag { bf16x16 v; uint4 u[2]; };
__device__ __forceinline__ bf16x16 load_bfrag(const __bf16* row_base, int byte_off) {
  BFrag r;
  const char* p = (const char*)row_base + byte_off;
  r.u[0] = *(const uint4*)(p);
  r.u[1] = *(const uint4*)(p + 16);
  return r.v;
}

// WMMA A-fragment (16x32 bf16) from a row-major bf16 LDS tile:
// lane ln+16*lg holds row ln; elem e -> k=16lg+e
union AFrag { bf16x16 v; unsigned long long q[4]; };
__device__ __forceinline__ bf16x16 load_afrag(const __bf16* base, int ln, int lg,
                                              int stride_bf16) {
  AFrag r;
  const char* p = (const char*)(base + ln * stride_bf16);
  r.q[0] = *(const unsigned long long*)(p + 32 * lg);
  r.q[1] = *(const unsigned long long*)(p + 32 * lg + 8);
  r.q[2] = *(const unsigned long long*)(p + 32 * lg + 16);
  r.q[3] = *(const unsigned long long*)(p + 32 * lg + 24);
  return r.v;
}

__global__ __launch_bounds__(128, 2)
void attn_postmask_kernel(const float* __restrict__ Q, const float* __restrict__ K,
                          const float* __restrict__ V, const float* __restrict__ MSK,
                          float* __restrict__ OUT) {
  __shared__ __align__(16) __bf16 sKb[kKT * kKSb];      // 32x64 K tile (bf16)
  __shared__ __align__(16) __bf16 sVTb[kDH * kVSb];     // 64x32 V^T tile (bf16)
  __shared__ __align__(16) __bf16 sP[4 * 16 * kPS];     // per-wave P staging (16x32)
  __shared__ __align__(16) __bf16 sM[64 * kPS];         // 64x32 mask tile (bf16)

  const int h    = blockIdx.y;
  const int tid  = threadIdx.x;
  const int wave = tid >> 5;
  const int lane = tid & 31;
  const int lg   = lane >> 4;
  const int ln   = lane & 15;
  const int q0wg = blockIdx.x * 64;
  const int q0   = q0wg + wave * 16;

  const float* Qh = Q + (size_t)h * kLQ * kDH;
  const float* Kh = K + (size_t)h * kLK * kDH;
  const float* Vh = V + (size_t)h * kLK * kDH;

  // Q A-fragments with softmax scale 1/sqrt(64) folded in
  bf16x16 aq0, aq1;
  {
    const float* qr = Qh + (size_t)(q0 + ln) * kDH;
    const float sc = 0.125f;
#pragma unroll
    for (int e = 0; e < 8; ++e) {
      aq0[e]     = (__bf16)(sc * qr[      16 * lg + e]);
      aq0[e + 8] = (__bf16)(sc * qr[ 8 +  16 * lg + e]);
      aq1[e]     = (__bf16)(sc * qr[32 +  16 * lg + e]);
      aq1[e + 8] = (__bf16)(sc * qr[40 +  16 * lg + e]);
    }
  }

  // Online-softmax state; lane's C-tile VGPR j corresponds to row (j + 8*lg)
  float m[8], l[8];
#pragma unroll
  for (int j = 0; j < 8; ++j) { m[j] = -3.0e38f; l[j] = 0.0f; }

  f32x8 osoft[4] = {};   // softmax(S)@V, rescaled online, normalized by 1/l at end
  f32x8 omask[4] = {};   // mask@V, never rescaled (mask is added AFTER softmax)
  __bf16* pst = &sP[wave * 16 * kPS];

  // ============== single streaming pass over K/V/mask ==============
  for (int kt = 0; kt < kLK; kt += kKT) {
    __syncthreads();
#pragma unroll
    for (int i = 0; i < 4; ++i) {
      int f = tid + 128 * i;
      // K tile: 32 rows x 64 -> bf16 row-major
      int krow = f >> 4, kc4 = f & 15;
      f32x4 kv = *(const f32x4*)(Kh + (size_t)(kt + krow) * kDH + kc4 * 4);
      *(bf16x4*)(&sKb[krow * kKSb + kc4 * 4]) = __builtin_convertvector(kv, bf16x4);
      // V tile: 32 rows x 64 -> bf16 transposed (64x32)
      f32x4 vv = *(const f32x4*)(Vh + (size_t)(kt + krow) * kDH + kc4 * 4);
      bf16x4 vb = __builtin_convertvector(vv, bf16x4);
      sVTb[(kc4 * 4 + 0) * kVSb + krow] = vb[0];
      sVTb[(kc4 * 4 + 1) * kVSb + krow] = vb[1];
      sVTb[(kc4 * 4 + 2) * kVSb + krow] = vb[2];
      sVTb[(kc4 * 4 + 3) * kVSb + krow] = vb[3];
      // mask tile: 64 q-rows x 32 k-cols -> bf16 row-major
      int mrow = f >> 3, mc4 = f & 7;
      f32x4 mv = *(const f32x4*)(MSK + (size_t)(q0wg + mrow) * kLK + kt + mc4 * 4);
      *(bf16x4*)(&sM[mrow * kPS + mc4 * 4]) = __builtin_convertvector(mv, bf16x4);
      // prefetch next K/V tile into cache while this iteration computes
      if (kt + kKT < kLK) {
        __builtin_prefetch(Kh + (size_t)(kt + kKT + krow) * kDH + kc4 * 4, 0, 3);
        __builtin_prefetch(Vh + (size_t)(kt + kKT + krow) * kDH + kc4 * 4, 0, 3);
      }
    }
    __syncthreads();

    // S = (Q*scale) K^T : two 16x16 tiles over 32 k-columns
    f32x8 s0, s1;
    {
      const __bf16* kr0 = &sKb[ln * kKSb];
      const __bf16* kr1 = &sKb[(16 + ln) * kKSb];
      f32x8 a = {};
      a = WMMA_BF16(aq0, load_bfrag(kr0, 32 * lg), a);
      a = WMMA_BF16(aq1, load_bfrag(kr0, 64 + 32 * lg), a);
      s0 = a;
      f32x8 b = {};
      b = WMMA_BF16(aq0, load_bfrag(kr1, 32 * lg), b);
      b = WMMA_BF16(aq1, load_bfrag(kr1, 64 + 32 * lg), b);
      s1 = b;
    }

    // online softmax update + stage P = exp(s - m_new) as bf16
    float rsc[8];
#pragma unroll
    for (int j = 0; j < 8; ++j) {
      float v0 = s0[j], v1 = s1[j];
      float mx = red_max16(VMAXF(v0, v1));
      float mn = VMAXF(m[j], mx);
      float p0 = __expf(v0 - mn), p1 = __expf(v1 - mn);
      float ts = red_sum16(p0 + p1);
      float r  = __expf(m[j] - mn);
      l[j] = l[j] * r + ts;
      m[j] = mn;
      rsc[j] = r;
      int row = j + 8 * lg;
      pst[row * kPS + ln]      = (__bf16)p0;
      pst[row * kPS + 16 + ln] = (__bf16)p1;
    }

    // rescale softmax accumulator by exp(m_old - m_new)
#pragma unroll
    for (int dc = 0; dc < 4; ++dc)
#pragma unroll
      for (int j = 0; j < 8; ++j) osoft[dc][j] *= rsc[j];

    // LDS stores above feed loads below (same wave; DS in-order)
    asm volatile("s_wait_dscnt 0x0" ::: "memory");

    bf16x16 paf = load_afrag(pst, ln, lg, kPS);                 // P fragment
    bf16x16 maf = load_afrag(&sM[wave * 16 * kPS], ln, lg, kPS); // mask fragment

    // O_soft += P @ V ; O_mask += mask @ V   (V fragment shared)
#pragma unroll
    for (int dc = 0; dc < 4; ++dc) {
      bf16x16 bv = load_bfrag(&sVTb[(dc * 16 + ln) * kVSb], 32 * lg);
      osoft[dc] = WMMA_BF16(paf, bv, osoft[dc]);
      omask[dc] = WMMA_BF16(maf, bv, omask[dc]);
    }
  }

  // ---- epilogue: out = O_soft / l + O_mask
  float invl[8];
#pragma unroll
  for (int j = 0; j < 8; ++j) invl[j] = 1.0f / l[j];

#pragma unroll
  for (int dc = 0; dc < 4; ++dc) {
#pragma unroll
    for (int j = 0; j < 8; ++j) {
      int row = q0 + j + 8 * lg;
      OUT[((size_t)h * kLQ + row) * kDH + dc * 16 + ln] =
          osoft[dc][j] * invl[j] + omask[dc][j];
    }
  }
}

extern "C" void kernel_launch(void* const* d_in, const int* in_sizes, int n_in,
                              void* d_out, int out_size, void* d_ws, size_t ws_size,
                              hipStream_t stream) {
  const float* q   = (const float*)d_in[0];
  const float* k   = (const float*)d_in[1];
  const float* v   = (const float*)d_in[2];
  const float* msk = (const float*)d_in[3];
  float* out = (float*)d_out;

  dim3 grid(kLQ / 64, kNH);   // 32 q-tiles x 16 heads; 4 waves/block, 16 q-rows/wave
  attn_postmask_kernel<<<grid, 128, 0, stream>>>(q, k, v, msk, out);
}
